// Matcher_5274219839805
// MI455X (gfx1250) — matrix operand order repaired and checked by reading, not compiled, not run on hardware
//
#include <hip/hip_runtime.h>
#include <hip/hip_bf16.h>

typedef __attribute__((ext_vector_type(16))) _Float16 v16h;
typedef __attribute__((ext_vector_type(8)))  _Float16 v8h;
typedef __attribute__((ext_vector_type(8)))  float    v8f;

#define BB    8
#define NN    4096
#define M1P   819
#define M2P   204
#define KNEI  64
#define KPOS  3072
#define DETK  512
#define BIGF  3.0e38f

__device__ __forceinline__ v8f wmma_f16(v16h a, v16h b, v8f c) {
  return __builtin_amdgcn_wmma_f32_16x16x32_f16(false, a, false, b, (short)0, c, false, false);
}

// A-fragment: two contiguous 8-half runs at p and p+16 (16B aligned)
__device__ __forceinline__ v16h ld_a_h16(const _Float16* p) {
  v8h lo = *(const v8h*)p;
  v8h hi = *(const v8h*)(p + 16);
  return __builtin_shufflevector(lo, hi, 0,1,2,3,4,5,6,7,8,9,10,11,12,13,14,15);
}
// B-fragment from transposed LDS weights: two contiguous 8-half runs at p and p+16 (16B aligned)
__device__ __forceinline__ v16h ld_b_h16(const _Float16* p) {
  v8h lo = *(const v8h*)p;
  v8h hi = *(const v8h*)(p + 16);
  return __builtin_shufflevector(lo, hi, 0,1,2,3,4,5,6,7,8,9,10,11,12,13,14,15);
}
// A-fragment from global f32 X (row stride padded): two runs of 8 floats
__device__ __forceinline__ v16h ld_a_f32(const float* p) {
  float4 r0 = *(const float4*)(p);
  float4 r1 = *(const float4*)(p + 4);
  float4 r2 = *(const float4*)(p + 16);
  float4 r3 = *(const float4*)(p + 20);
  v16h a;
  a[0]=(_Float16)r0.x;  a[1]=(_Float16)r0.y;  a[2]=(_Float16)r0.z;  a[3]=(_Float16)r0.w;
  a[4]=(_Float16)r1.x;  a[5]=(_Float16)r1.y;  a[6]=(_Float16)r1.z;  a[7]=(_Float16)r1.w;
  a[8]=(_Float16)r2.x;  a[9]=(_Float16)r2.y;  a[10]=(_Float16)r2.z; a[11]=(_Float16)r2.w;
  a[12]=(_Float16)r3.x; a[13]=(_Float16)r3.y; a[14]=(_Float16)r3.z; a[15]=(_Float16)r3.w;
  return a;
}

// ---------------------------------------------------------------------------
// Generic WMMA linear: Y[M,Nout] = act(X[M,Kp-padded] @ W[Kin,Nout] + bias)
// Weights staged transposed+padded in LDS; one 16x16 tile per wave.
// ---------------------------------------------------------------------------
__global__ __launch_bounds__(256) void wmma_linear_kernel(
    const float* __restrict__ X, int Kp, const float* __restrict__ W,
    const float* __restrict__ bias, float* __restrict__ Y,
    int M, int Kin, int Nout, int relu)
{
  __shared__ __align__(16) _Float16 Wt[128 * 64];   // [Nout][Kp]
  __shared__ float Bs[64];
  int t = threadIdx.x;
  for (int idx = t; idx < Nout * Kp; idx += 256) {
    int n = idx / Kp, k = idx % Kp;
    Wt[idx] = (_Float16)((k < Kin) ? W[(size_t)k * Nout + n] : 0.f);
  }
  for (int n = t; n < Nout; n += 256) Bs[n] = bias[n];
  __syncthreads();

  int wave = t >> 5, lane = t & 31;
  int ntiles = Nout >> 4;
  int mtiles = (M + 15) >> 4;
  int tile = blockIdx.x * 8 + wave;
  if (tile >= mtiles * ntiles) return;
  int mt = tile / ntiles, nt = tile % ntiles;
  int rowA = mt * 16 + (lane & 15);
  int kh   = lane >> 4;
  int colB = nt * 16 + (lane & 15);
  v8f c = {};
  int kchunks = Kp >> 5;
  const float*    ap = X  + (size_t)rowA * Kp + kh * 8;
  const _Float16* bp = Wt + colB * Kp + kh * 8;
  for (int kc = 0; kc < kchunks; ++kc) {
    v16h a  = ld_a_f32(ap + kc * 32);
    v16h bm = ld_b_h16(bp + kc * 32);
    c = wmma_f16(a, bm, c);
  }
  float bb = Bs[colB];
#pragma unroll
  for (int v = 0; v < 8; ++v) {
    int row = mt * 16 + v + kh * 8;
    if (row < M) {
      float y = c[v] + bb;
      if (relu) y = fmaxf(y, 0.f);
      Y[(size_t)row * Nout + colB] = y;
    }
  }
}

// ---------------------------------------------------------------------------
// Fused PointNetConv: ball query (k=64 nearest-within-r) + 2-layer MLP via
// WMMA (weights staged in LDS) + masked max aggregation. Block per center.
// ---------------------------------------------------------------------------
__global__ __launch_bounds__(256) void pn_conv_kernel(
    const float* __restrict__ xsrc, int finx,
    const float* __restrict__ psrc, const float* __restrict__ pdst,
    int n, int m, float r2,
    const float* __restrict__ W1, const float* __restrict__ b1,   // [kin][32]
    const float* __restrict__ W2, const float* __restrict__ b2,   // [32][fout]
    int fout, float* __restrict__ out)
{
  __shared__ float    d2s[NN];
  __shared__ int      selIdx[KNEI];
  __shared__ int      selValid[KNEI];
  __shared__ __align__(16) _Float16 Xs[KNEI * 64];
  __shared__ __align__(16) _Float16 Hs[KNEI * 32];
  __shared__ __align__(16) _Float16 W1t[32 * 64];    // [32][kinPad=64]
  __shared__ __align__(16) _Float16 W2t[64 * 32];    // [fout][32]
  __shared__ float    b1s[32];
  __shared__ float    b2s[64];
  __shared__ float    redv[256];
  __shared__ int      redi[256];
  __shared__ float    partial[16 * 32];

  int b  = blockIdx.x / m;
  int cm = blockIdx.x % m;
  int t  = threadIdx.x;
  int kin = finx + 3;
  const float* ps = psrc + (size_t)b * n * 3;
  const float* pd = pdst + ((size_t)b * m + cm) * 3;
  float cx = pd[0], cy = pd[1], cz = pd[2];

  // stage weights (transposed, zero-padded) + biases
  for (int idx = t; idx < 32 * 64; idx += 256) {
    int nn = idx >> 6, k = idx & 63;
    W1t[idx] = (_Float16)((k < kin) ? W1[(size_t)k * 32 + nn] : 0.f);
  }
  for (int idx = t; idx < fout * 32; idx += 256) {
    int nn = idx >> 5, k = idx & 31;
    W2t[idx] = (_Float16)W2[(size_t)k * fout + nn];
  }
  if (t < 32) b1s[t] = b1[t];
  if (t < fout) b2s[t] = b2[t];

  for (int i = t; i < n; i += 256) {
    float dx = ps[i * 3] - cx, dy = ps[i * 3 + 1] - cy, dz = ps[i * 3 + 2] - cz;
    d2s[i] = dx * dx + dy * dy + dz * dz;
  }
  __syncthreads();

  // k=64 nearest (iterative argmin, stable ties)
  for (int kq = 0; kq < KNEI; ++kq) {
    float bv = BIGF; int bi = NN;
    for (int i = t; i < n; i += 256) {
      float v = d2s[i];
      if (v < bv || (v == bv && i < bi)) { bv = v; bi = i; }
    }
    redv[t] = bv; redi[t] = bi;
    __syncthreads();
    for (int s = 128; s > 0; s >>= 1) {
      if (t < s) {
        if (redv[t + s] < redv[t] || (redv[t + s] == redv[t] && redi[t + s] < redi[t])) {
          redv[t] = redv[t + s]; redi[t] = redi[t + s];
        }
      }
      __syncthreads();
    }
    if (t == 0) {
      int ii = redi[0];
      selIdx[kq]   = ii;
      selValid[kq] = (redv[0] < r2) ? 1 : 0;
      if (ii < n) d2s[ii] = BIGF;
    }
    __syncthreads();
  }

  // build message matrix Xs [64 x 64] (f16, zero-padded)
  for (int e = t; e < KNEI * 64; e += 256) {
    int rr = e >> 6, col = e & 63;
    float val = 0.f;
    int si = selIdx[rr];
    if (si < n) {
      if (col < finx)      val = xsrc[((size_t)b * n + si) * finx + col];
      else if (col < kin)  { int d = col - finx; val = ps[si * 3 + d] - pd[d]; }
    }
    Xs[e] = (_Float16)val;
  }
  __syncthreads();

  int wave = t >> 5, lane = t & 31;

  // layer 1: H[64x32] = relu(Xs @ W1 + b1); 8 waves = 4 mtiles x 2 ntiles
  {
    int mt = wave & 3, nt = wave >> 2;
    int rowA = mt * 16 + (lane & 15);
    int kh   = lane >> 4;
    int colB = nt * 16 + (lane & 15);
    v8f c = {};
    int kchunks = (kin > 32) ? 2 : 1;
    const _Float16* ap = Xs  + rowA * 64 + kh * 8;
    const _Float16* bp = W1t + colB * 64 + kh * 8;
    for (int kc = 0; kc < kchunks; ++kc) {
      c = wmma_f16(ld_a_h16(ap + kc * 32), ld_b_h16(bp + kc * 32), c);
    }
    float bb = b1s[colB];
#pragma unroll
    for (int v = 0; v < 8; ++v) {
      int row = mt * 16 + v + kh * 8;
      Hs[row * 32 + colB] = (_Float16)fmaxf(c[v] + bb, 0.f);
    }
  }
  __syncthreads();

  // layer 2 + per-lane masked row max
  int ntiles2   = fout >> 4;        // 2 (sa1) or 4 (sa2)
  int ntilesTot = 4 * ntiles2;      // 8 or 16
  for (int tl = wave; tl < ntilesTot; tl += 8) {
    int mt = tl / ntiles2, nt = tl % ntiles2;
    int rowA = mt * 16 + (lane & 15);
    int kh   = lane >> 4;
    int colB = nt * 16 + (lane & 15);
    v16h a  = ld_a_h16(Hs + rowA * 32 + kh * 8);
    v16h bm = ld_b_h16(W2t + colB * 32 + kh * 8);
    v8f c = {};
    c = wmma_f16(a, bm, c);
    float bb = b2s[colB];
    float mx = -BIGF;
#pragma unroll
    for (int v = 0; v < 8; ++v) {
      int row = mt * 16 + v + kh * 8;
      float y = c[v] + bb;
      if (selValid[row]) mx = fmaxf(mx, y);
    }
    partial[tl * 32 + lane] = mx;
  }
  __syncthreads();

  for (int ci = t; ci < fout; ci += 256) {
    int nt = ci >> 4, cl = ci & 15;
    float mx = -BIGF;
    for (int mt = 0; mt < 4; ++mt)
      for (int half = 0; half < 2; ++half)
        mx = fmaxf(mx, partial[(mt * ntiles2 + nt) * 32 + half * 16 + cl]);
    if (mx < -1.0e30f) mx = 0.f;   // all-invalid -> 0 (ref: where(isfinite,...,0))
    out[((size_t)b * m + cm) * fout + ci] = mx;
  }
}

// ---------------------------------------------------------------------------
// Farthest point sampling: one block per batch; pos [B][n][3]
// ---------------------------------------------------------------------------
__global__ __launch_bounds__(256) void fps_kernel(
    const float* __restrict__ pos, int n, int m, int* __restrict__ outIdx)
{
  __shared__ float mind[NN];
  __shared__ float redv[256];
  __shared__ int   redi[256];
  __shared__ int   lastS;
  int b = blockIdx.x, t = threadIdx.x;
  const float* p = pos + (size_t)b * n * 3;
  int* oi = outIdx + (size_t)b * m;
  for (int i = t; i < n; i += 256) mind[i] = BIGF;
  if (t == 0) { oi[0] = 0; lastS = 0; }
  __syncthreads();
  for (int it = 1; it < m; ++it) {
    int last = lastS;
    float lx = p[last * 3], ly = p[last * 3 + 1], lz = p[last * 3 + 2];
    float bv = -1.f; int bi = 0;
    for (int i = t; i < n; i += 256) {
      float dx = p[i * 3] - lx, dy = p[i * 3 + 1] - ly, dz = p[i * 3 + 2] - lz;
      float d = dx * dx + dy * dy + dz * dz;
      float mv = fminf(mind[i], d);
      mind[i] = mv;
      if (mv > bv) { bv = mv; bi = i; }
    }
    redv[t] = bv; redi[t] = bi;
    __syncthreads();
    for (int s = 128; s > 0; s >>= 1) {
      if (t < s) {
        if (redv[t + s] > redv[t] || (redv[t + s] == redv[t] && redi[t + s] < redi[t])) {
          redv[t] = redv[t + s]; redi[t] = redi[t + s];
        }
      }
      __syncthreads();
    }
    if (t == 0) { lastS = redi[0]; oi[it] = redi[0]; }
    __syncthreads();
  }
}

// ---------------------------------------------------------------------------
// Small glue kernels
// ---------------------------------------------------------------------------
__global__ void prep_kernel(const float* __restrict__ pos, float* __restrict__ loc,
                            float* __restrict__ f0, int total)
{
  int r = blockIdx.x * blockDim.x + threadIdx.x;
  if (r >= total) return;
  loc[r * 3 + 0] = pos[r * 6 + 0];
  loc[r * 3 + 1] = pos[r * 6 + 1];
  loc[r * 3 + 2] = pos[r * 6 + 2];
  f0[r] = sinf(0.1f * pos[r * 6 + 1]);
}

__global__ void gather3_kernel(const float* __restrict__ src, const int* __restrict__ idx,
                               float* __restrict__ dst, int nsrc, int m)
{
  int r = blockIdx.x * blockDim.x + threadIdx.x;
  int total = BB * m * 3;
  if (r >= total) return;
  int b = r / (m * 3), rem = r % (m * 3), j = rem / 3, d = rem % 3;
  dst[r] = src[((size_t)b * nsrc + idx[(size_t)b * m + j]) * 3 + d];
}

// concat with zero-padding to Kp columns
__global__ void concat2_pad_kernel(const float* __restrict__ A, int Fa,
                                   const float* __restrict__ Bp, int Fb,
                                   float* __restrict__ out, int rows, int Kp)
{
  int e = blockIdx.x * blockDim.x + threadIdx.x;
  if (e >= rows * Kp) return;
  int r = e / Kp, c = e % Kp;
  float v = 0.f;
  if (c < Fa) v = A[(size_t)r * Fa + c];
  else if (c < Fa + Fb) v = Bp[(size_t)r * Fb + (c - Fa)];
  out[e] = v;
}

__global__ void bcastcat_kernel(const float* __restrict__ x3, const float* __restrict__ x2,
                                float* __restrict__ out, int m2)
{
  int e = blockIdx.x * blockDim.x + threadIdx.x;
  int total = BB * m2 * 128;
  if (e >= total) return;
  int r = e / 128, c = e % 128, b = r / m2;
  out[e] = (c < 64) ? x3[b * 64 + c] : x2[(size_t)r * 64 + (c - 64)];
}

__global__ void rowmax_kernel(const float* __restrict__ X, int m, int F, float* __restrict__ out)
{
  int b = blockIdx.x;
  for (int f = threadIdx.x; f < F; f += blockDim.x) {
    float mx = -BIGF;
    for (int r = 0; r < m; ++r) mx = fmaxf(mx, X[((size_t)b * m + r) * F + f]);
    out[b * F + f] = mx;
  }
}

__global__ void knn3_kernel(const float* __restrict__ xsrc, const float* __restrict__ psrc,
                            const float* __restrict__ ptgt, int nsrc, int ntgt, int F,
                            float* __restrict__ out)
{
  int r = blockIdx.x * blockDim.x + threadIdx.x;
  if (r >= BB * ntgt) return;
  int b = r / ntgt;
  const float* pt = ptgt + (size_t)r * 3;
  float tx = pt[0], ty = pt[1], tz = pt[2];
  float bd0 = BIGF, bd1 = BIGF, bd2 = BIGF;
  int   bi0 = 0, bi1 = 0, bi2 = 0;
  const float* ps = psrc + (size_t)b * nsrc * 3;
  for (int i = 0; i < nsrc; ++i) {
    float dx = ps[i * 3] - tx, dy = ps[i * 3 + 1] - ty, dz = ps[i * 3 + 2] - tz;
    float d = dx * dx + dy * dy + dz * dz;
    if (d < bd0) { bd2 = bd1; bi2 = bi1; bd1 = bd0; bi1 = bi0; bd0 = d; bi0 = i; }
    else if (d < bd1) { bd2 = bd1; bi2 = bi1; bd1 = d; bi1 = i; }
    else if (d < bd2) { bd2 = d; bi2 = i; }
  }
  float w0 = 1.f / fmaxf(bd0, 1e-16f);
  float w1 = 1.f / fmaxf(bd1, 1e-16f);
  float w2 = 1.f / fmaxf(bd2, 1e-16f);
  float wsum = w0 + w1 + w2;
  const float* xs = xsrc + (size_t)b * nsrc * F;
  float* o = out + (size_t)r * F;
  for (int f = 0; f < F; ++f)
    o[f] = (w0 * xs[(size_t)bi0 * F + f] + w1 * xs[(size_t)bi1 * F + f] +
            w2 * xs[(size_t)bi2 * F + f]) / wsum;
}

__global__ void detector_kernel(const float* __restrict__ dense,
                                const float* __restrict__ W1, const float* __restrict__ B1,
                                const float* __restrict__ W2, const float* __restrict__ B2,
                                const float* __restrict__ WL, const float* __restrict__ BL,
                                float* __restrict__ wout, int total)
{
  int r = blockIdx.x * blockDim.x + threadIdx.x;
  if (r >= total) return;
  const float* x = dense + (size_t)r * 32;
  float h[16];
  for (int o = 0; o < 16; ++o) {
    float s = B1[o];
    for (int i = 0; i < 32; ++i) s += x[i] * W1[i * 16 + o];
    h[o] = fmaxf(s, 0.f);
  }
  float g[8];
  for (int o = 0; o < 8; ++o) {
    float s = B2[o];
    for (int i = 0; i < 16; ++i) s += h[i] * W2[i * 8 + o];
    g[o] = s;
  }
  float s = BL[0];
  for (int i = 0; i < 8; ++i) s += g[i] * WL[i];
  wout[r] = (s > 20.f) ? s : log1pf(expf(s));
}

__global__ void dc_kernel(const float* __restrict__ loc, float* __restrict__ dc, int total)
{
  int r = blockIdx.x * blockDim.x + threadIdx.x;
  if (r >= total) return;
  float x = loc[r * 3], y = loc[r * 3 + 1];
  dc[r] = sqrtf(x * x + y * y);
}

// stable rank-select of the KPOS closest-to-center points (matches top_k order)
__global__ void rank_kernel(const float* __restrict__ dc, const float* __restrict__ loc,
                            int* __restrict__ sel, float* __restrict__ selpos)
{
  int r = blockIdx.x * blockDim.x + threadIdx.x;
  if (r >= BB * NN) return;
  int b = r / NN, i = r % NN;
  const float* d = dc + (size_t)b * NN;
  float di = d[i];
  int cnt = 0;
  for (int j = 0; j < NN; ++j) {
    float dj = d[j];
    cnt += (dj < di) || (dj == di && j < i);
  }
  if (cnt < KPOS) {
    sel[(size_t)b * KPOS + cnt] = i;
    selpos[((size_t)b * KPOS + cnt) * 3 + 0] = loc[(size_t)r * 3 + 0];
    selpos[((size_t)b * KPOS + cnt) * 3 + 1] = loc[(size_t)r * 3 + 1];
    selpos[((size_t)b * KPOS + cnt) * 3 + 2] = loc[(size_t)r * 3 + 2];
  }
}

__global__ void keys_kernel(const int* __restrict__ sel, const int* __restrict__ keyloc,
                            const float* __restrict__ wdet, const float* __restrict__ pos1,
                            const float* __restrict__ center,
                            float* __restrict__ keyxyz, int* __restrict__ anch0,
                            float* __restrict__ out_kw, float* __restrict__ out_kidx)
{
  int r = blockIdx.x * blockDim.x + threadIdx.x;
  if (r >= BB * DETK) return;
  int b = r / DETK;
  int kl = keyloc[r];
  int o  = sel[(size_t)b * KPOS + kl];
  float ax = pos1[((size_t)b * NN + o) * 6 + 3];
  float ay = pos1[((size_t)b * NN + o) * 6 + 4];
  float az = pos1[((size_t)b * NN + o) * 6 + 5];
  keyxyz[r * 3 + 0] = ax; keyxyz[r * 3 + 1] = ay; keyxyz[r * 3 + 2] = az;
  float c1x = center[b * 6 + 0], c1y = center[b * 6 + 1];
  float c2x = center[b * 6 + 3], c2y = center[b * 6 + 4];
  float dc1 = sqrtf((ax - c1x) * (ax - c1x) + (ay - c1y) * (ay - c1y));
  float dc2 = sqrtf((ax - c2x) * (ax - c2x) + (ay - c2y) * (ay - c2y));
  anch0[r] = (dc1 < 40.f && dc2 < 40.f) ? 1 : 0;
  out_kw[r]   = wdet[(size_t)b * NN + o];
  out_kidx[r] = (float)o;
}

__global__ __launch_bounds__(256) void match_kernel(
    const float* __restrict__ keyxyz, const float* __restrict__ pos2,
    const int* __restrict__ anch0, float* __restrict__ out_mind,
    float* __restrict__ out_mini, float* __restrict__ out_anch)
{
  __shared__ float rv[256];
  __shared__ int   ri[256];
  int bk = blockIdx.x;         // b*512 + k
  int b  = bk / DETK;
  int t  = threadIdx.x;
  float kx = keyxyz[bk * 3], ky = keyxyz[bk * 3 + 1], kz = keyxyz[bk * 3 + 2];
  const float* p2 = pos2 + (size_t)b * NN * 6;
  float bv = BIGF; int bi = 0;
  for (int j = t; j < NN; j += 256) {
    float dx = p2[j * 6 + 3] - kx, dy = p2[j * 6 + 4] - ky, dz = p2[j * 6 + 5] - kz;
    float d = dx * dx + dy * dy + dz * dz;
    if (d < bv || (d == bv && j < bi)) { bv = d; bi = j; }
  }
  rv[t] = bv; ri[t] = bi;
  __syncthreads();
  for (int s = 128; s > 0; s >>= 1) {
    if (t < s) {
      if (rv[t + s] < rv[t] || (rv[t + s] == rv[t] && ri[t + s] < ri[t])) {
        rv[t] = rv[t + s]; ri[t] = ri[t + s];
      }
    }
    __syncthreads();
  }
  if (t == 0) {
    float md = sqrtf(fmaxf(rv[0], 1e-12f));
    out_mind[bk] = md;
    out_mini[bk] = (float)ri[0];
    out_anch[bk] = (anch0[bk] && md < 1.0f) ? 1.f : 0.f;
  }
}

// ---------------------------------------------------------------------------
// Driver
// ---------------------------------------------------------------------------
static float* allocF(char*& p, size_t n) {
  float* r = (float*)p;
  p += (((n + 4096) * sizeof(float)) + 255) & ~((size_t)255);  // +slack for tile overread
  return r;
}
static int* allocI(char*& p, size_t n) {
  int* r = (int*)p;
  p += (((n + 256) * sizeof(int)) + 255) & ~((size_t)255);
  return r;
}

extern "C" void kernel_launch(void* const* d_in, const int* in_sizes, int n_in,
                              void* d_out, int out_size, void* d_ws, size_t ws_size,
                              hipStream_t stream) {
  (void)in_sizes; (void)n_in; (void)out_size; (void)ws_size;

  const float* pos1   = (const float*)d_in[0];
  const float* pos2   = (const float*)d_in[1];
  const float* center = (const float*)d_in[2];
  const float* detW1  = (const float*)d_in[3];
  const float* detB1  = (const float*)d_in[4];
  const float* detW2  = (const float*)d_in[5];
  const float* detB2  = (const float*)d_in[6];
  const float* dlW    = (const float*)d_in[7];
  const float* dlB    = (const float*)d_in[8];
  const float* fp1W1  = (const float*)d_in[9];
  const float* fp1B1  = (const float*)d_in[10];
  const float* fp1W2  = (const float*)d_in[11];
  const float* fp1B2  = (const float*)d_in[12];
  const float* fp1W3  = (const float*)d_in[13];
  const float* fp1B3  = (const float*)d_in[14];
  const float* fp2W1  = (const float*)d_in[15];
  const float* fp2B1  = (const float*)d_in[16];
  const float* fp2W2  = (const float*)d_in[17];
  const float* fp2B2  = (const float*)d_in[18];
  const float* fp3W1  = (const float*)d_in[19];
  const float* fp3B1  = (const float*)d_in[20];
  const float* fp3W2  = (const float*)d_in[21];
  const float* fp3B2  = (const float*)d_in[22];
  const float* hW1    = (const float*)d_in[23];
  const float* hB1    = (const float*)d_in[24];
  const float* hW2    = (const float*)d_in[25];
  const float* hB2    = (const float*)d_in[26];
  const float* hW3    = (const float*)d_in[27];
  const float* hB3    = (const float*)d_in[28];
  const float* sa1W1  = (const float*)d_in[29];
  const float* sa1B1  = (const float*)d_in[30];
  const float* sa1W2  = (const float*)d_in[31];
  const float* sa1B2  = (const float*)d_in[32];
  const float* sa2W1  = (const float*)d_in[33];
  const float* sa2B1  = (const float*)d_in[34];
  const float* sa2W2  = (const float*)d_in[35];
  const float* sa2B2  = (const float*)d_in[36];
  const float* sa3W1  = (const float*)d_in[37];
  const float* sa3B1  = (const float*)d_in[38];
  const float* sa3W2  = (const float*)d_in[39];
  const float* sa3B2  = (const float*)d_in[40];

  float* outf       = (float*)d_out;
  float* dense_src  = outf;
  float* dense_tgt  = outf + (size_t)BB * NN * 32;
  float* o_kw       = outf + (size_t)2 * BB * NN * 32;
  float* o_kidx     = o_kw   + (size_t)BB * DETK;
  float* o_mind     = o_kidx + (size_t)BB * DETK;
  float* o_mini     = o_mind + (size_t)BB * DETK;
  float* o_anch     = o_mini + (size_t)BB * DETK;

  char* wp = (char*)d_ws;
  float* loc    = allocF(wp, (size_t)BB * NN * 3);
  float* f0     = allocF(wp, (size_t)BB * NN);
  int*   i1     = allocI(wp, (size_t)BB * M1P);
  float* pos1s  = allocF(wp, (size_t)BB * M1P * 3);
  float* x1     = allocF(wp, (size_t)BB * M1P * 32);
  int*   i2     = allocI(wp, (size_t)BB * M2P);
  float* pos2s  = allocF(wp, (size_t)BB * M2P * 3);
  float* x2     = allocF(wp, (size_t)BB * M2P * 64);
  float* sa3cat = allocF(wp, (size_t)BB * M2P * 96);   // 67 -> pad 96
  float* sa3h   = allocF(wp, (size_t)BB * M2P * 64);
  float* sa3o   = allocF(wp, (size_t)BB * M2P * 64);
  float* x3     = allocF(wp, (size_t)BB * 64);
  float* fp3cat = allocF(wp, (size_t)BB * M2P * 128);
  float* fp3h   = allocF(wp, (size_t)BB * M2P * 64);
  float* fp3o   = allocF(wp, (size_t)BB * M2P * 64);
  float* itp1   = allocF(wp, (size_t)BB * M1P * 64);
  float* fp2cat = allocF(wp, (size_t)BB * M1P * 96);
  float* MA     = allocF(wp, (size_t)BB * M1P * 32);
  float* MBuf   = allocF(wp, (size_t)BB * M1P * 32);
  float* itp0   = allocF(wp, (size_t)BB * NN * 32);
  float* fp1cat = allocF(wp, (size_t)BB * NN * 64);    // 33 -> pad 64
  float* GA     = allocF(wp, (size_t)BB * NN * 32);
  float* GB     = allocF(wp, (size_t)BB * NN * 32);
  float* wdet   = allocF(wp, (size_t)BB * NN);
  float* dc     = allocF(wp, (size_t)BB * NN);
  int*   sel    = allocI(wp, (size_t)BB * KPOS);
  float* selpos = allocF(wp, (size_t)BB * KPOS * 3);
  int*   keyloc = allocI(wp, (size_t)BB * DETK);
  float* keyxyz = allocF(wp, (size_t)BB * DETK * 3);
  int*   anch0  = allocI(wp, (size_t)BB * DETK);

  const int TPB = 256;
  #define GRID(n) dim3((((n) + TPB - 1) / TPB))
  #define LIN_GRID(M, Nout) dim3(((((M) + 15) / 16) * ((Nout) / 16) + 7) / 8)

  for (int s = 0; s < 2; ++s) {
    const float* pos = (s == 0) ? pos1 : pos2;
    float* dense = (s == 0) ? dense_src : dense_tgt;

    prep_kernel<<<GRID(BB * NN), TPB, 0, stream>>>(pos, loc, f0, BB * NN);

    fps_kernel<<<BB, TPB, 0, stream>>>(loc, NN, M1P, i1);
    gather3_kernel<<<GRID(BB * M1P * 3), TPB, 0, stream>>>(loc, i1, pos1s, NN, M1P);
    pn_conv_kernel<<<BB * M1P, TPB, 0, stream>>>(f0, 1, loc, pos1s, NN, M1P, 64.f,
                                                 sa1W1, sa1B1, sa1W2, sa1B2, 32, x1);

    fps_kernel<<<BB, TPB, 0, stream>>>(pos1s, M1P, M2P, i2);
    gather3_kernel<<<GRID(BB * M2P * 3), TPB, 0, stream>>>(pos1s, i2, pos2s, M1P, M2P);
    pn_conv_kernel<<<BB * M2P, TPB, 0, stream>>>(x1, 32, pos1s, pos2s, M1P, M2P, 256.f,
                                                 sa2W1, sa2B1, sa2W2, sa2B2, 64, x2);

    // sa3 + global max pool
    concat2_pad_kernel<<<GRID(BB * M2P * 96), TPB, 0, stream>>>(x2, 64, pos2s, 3, sa3cat, BB * M2P, 96);
    wmma_linear_kernel<<<LIN_GRID(BB * M2P, 64), TPB, 0, stream>>>(sa3cat, 96, sa3W1, sa3B1, sa3h, BB * M2P, 67, 64, 1);
    wmma_linear_kernel<<<LIN_GRID(BB * M2P, 64), TPB, 0, stream>>>(sa3h, 64, sa3W2, sa3B2, sa3o, BB * M2P, 64, 64, 0);
    rowmax_kernel<<<BB, TPB, 0, stream>>>(sa3o, M2P, 64, x3);

    // fp3
    bcastcat_kernel<<<GRID(BB * M2P * 128), TPB, 0, stream>>>(x3, x2, fp3cat, M2P);
    wmma_linear_kernel<<<LIN_GRID(BB * M2P, 64), TPB, 0, stream>>>(fp3cat, 128, fp3W1, fp3B1, fp3h, BB * M2P, 128, 64, 1);
    wmma_linear_kernel<<<LIN_GRID(BB * M2P, 64), TPB, 0, stream>>>(fp3h, 64, fp3W2, fp3B2, fp3o, BB * M2P, 64, 64, 0);

    // fp2
    knn3_kernel<<<GRID(BB * M1P), TPB, 0, stream>>>(fp3o, pos2s, pos1s, M2P, M1P, 64, itp1);
    concat2_pad_kernel<<<GRID(BB * M1P * 96), TPB, 0, stream>>>(itp1, 64, x1, 32, fp2cat, BB * M1P, 96);
    wmma_linear_kernel<<<LIN_GRID(BB * M1P, 32), TPB, 0, stream>>>(fp2cat, 96, fp2W1, fp2B1, MA, BB * M1P, 96, 32, 1);
    wmma_linear_kernel<<<LIN_GRID(BB * M1P, 32), TPB, 0, stream>>>(MA, 32, fp2W2, fp2B2, MBuf, BB * M1P, 32, 32, 0);

    // fp1
    knn3_kernel<<<GRID(BB * NN), TPB, 0, stream>>>(MBuf, pos1s, loc, M1P, NN, 32, itp0);
    concat2_pad_kernel<<<GRID(BB * NN * 64), TPB, 0, stream>>>(itp0, 32, f0, 1, fp1cat, BB * NN, 64);
    wmma_linear_kernel<<<LIN_GRID(BB * NN, 32), TPB, 0, stream>>>(fp1cat, 64, fp1W1, fp1B1, GA, BB * NN, 33, 32, 1);
    wmma_linear_kernel<<<LIN_GRID(BB * NN, 32), TPB, 0, stream>>>(GA, 32, fp1W2, fp1B2, GB, BB * NN, 32, 32, 1);
    wmma_linear_kernel<<<LIN_GRID(BB * NN, 32), TPB, 0, stream>>>(GB, 32, fp1W3, fp1B3, GA, BB * NN, 32, 32, 0);

    // head -> dense output
    wmma_linear_kernel<<<LIN_GRID(BB * NN, 32), TPB, 0, stream>>>(GA, 32, hW1, hB1, GB, BB * NN, 32, 32, 1);
    wmma_linear_kernel<<<LIN_GRID(BB * NN, 32), TPB, 0, stream>>>(GB, 32, hW2, hB2, GA, BB * NN, 32, 32, 1);
    wmma_linear_kernel<<<LIN_GRID(BB * NN, 32), TPB, 0, stream>>>(GA, 32, hW3, hB3, dense, BB * NN, 32, 32, 0);
  }

  // detector on dense_src with loc1 (recompute loc for submap 1)
  prep_kernel<<<GRID(BB * NN), TPB, 0, stream>>>(pos1, loc, f0, BB * NN);
  detector_kernel<<<GRID(BB * NN), TPB, 0, stream>>>(dense_src, detW1, detB1, detW2, detB2,
                                                     dlW, dlB, wdet, BB * NN);
  dc_kernel<<<GRID(BB * NN), TPB, 0, stream>>>(loc, dc, BB * NN);
  rank_kernel<<<GRID(BB * NN), TPB, 0, stream>>>(dc, loc, sel, selpos);
  fps_kernel<<<BB, TPB, 0, stream>>>(selpos, KPOS, DETK, keyloc);
  keys_kernel<<<GRID(BB * DETK), TPB, 0, stream>>>(sel, keyloc, wdet, pos1, center,
                                                   keyxyz, anch0, o_kw, o_kidx);
  match_kernel<<<BB * DETK, TPB, 0, stream>>>(keyxyz, pos2, anch0, o_mind, o_mini, o_anch);

  #undef GRID
  #undef LIN_GRID
}
